// EfficientVisionEncoder_51273319580026
// MI455X (gfx1250) — matrix operand order repaired and checked by reading, not compiled, not run on hardware
//
#include <hip/hip_runtime.h>
#include <math.h>
#include <stdint.h>

typedef __attribute__((ext_vector_type(16))) _Float16 v16h;
typedef __attribute__((ext_vector_type(8)))  _Float16 v8h;
typedef __attribute__((ext_vector_type(8)))  float    v8f;
typedef int v4i __attribute__((vector_size(16)));

#define HID   768
#define NH    12
#define HD    64
#define SEQ   64
#define BATCH 256
#define MTOT  (BATCH*SEQ)   // 16384 token rows

// Async global->LDS path (CDNA5 GLOBAL_LOAD_ASYNC_TO_LDS_B128), gated so the
// file compiles even if this toolchain lacks the builtins.
#if defined(__has_builtin)
#if __has_builtin(__builtin_amdgcn_global_load_async_to_lds_b128) && \
    __has_builtin(__builtin_amdgcn_s_wait_asynccnt)
#define GEMM_ASYNC 1
#endif
#endif
#ifndef GEMM_ASYNC
#define GEMM_ASYNC 0
#endif

// ---------------------------------------------------------------- helpers ---

__device__ inline float gelu_f(float x) {
  return 0.5f * x * (1.0f + erff(x * 0.70710678118654752440f));
}

__device__ inline v8f zero_v8f() {
  v8f z;
#pragma unroll
  for (int i = 0; i < 8; i++) z[i] = 0.f;
  return z;
}

// A-operand fragment (16x32 f16): lanes 0-15 = rows M; elems 0..15 -> K =
// lh*16..lh*16+15 (lh = lane>>4). `row` points at
// the K-chunk start of this lane's matrix row in LDS.
__device__ inline v16h load_frag_a(const _Float16* row, int lh) {
  v8h lo = *(const v8h*)(row + lh * 16);
  v8h hi = *(const v8h*)(row + lh * 16 + 8);
  v16h r;
#pragma unroll
  for (int i = 0; i < 8; i++) { r[i] = lo[i]; r[8 + i] = hi[i]; }
  return r;
}

// B-operand fragment (32x16 f16): lanes 0-15 = col N with K=0..15, lanes
// 16-31 same cols K=16..31. `row` = K-chunk start of this lane's (transposed)
// column in LDS, contiguous along K.
__device__ inline v16h load_frag_b(const _Float16* row, int lh) {
  v8h lo = *(const v8h*)(row + lh * 16);
  v8h hi = *(const v8h*)(row + lh * 16 + 8);
  v16h r;
#pragma unroll
  for (int i = 0; i < 8; i++) { r[i] = lo[i]; r[8 + i] = hi[i]; }
  return r;
}

#if GEMM_ASYNC
// Pointer types exactly matching the builtin: v4i in AS1 / AS3.
typedef __attribute__((address_space(1))) v4i* gld_ptr;
typedef __attribute__((address_space(3))) v4i* lds_ptr;
// One 16-byte per-lane async copy, global -> LDS (tracked by ASYNCcnt).
__device__ __forceinline__ void async16(const _Float16* g, _Float16* l) {
  __builtin_amdgcn_global_load_async_to_lds_b128(
      (gld_ptr)(uintptr_t)g, (lds_ptr)(unsigned)(uintptr_t)l, 0, 0);
}
#endif

// ---------------------------------------------------------------- zeroing ---

__global__ void zero_kernel(float* __restrict__ p, int n) {
  int i = blockIdx.x * blockDim.x + threadIdx.x;
  if (i < n) p[i] = 0.f;
}

// ----------------------------------------------------- weight transpose -----
// WT[n][k] = (f16) W[k][n].  32x32 LDS-tiled; K,N multiples of 32.
// Done once per GEMM (amortized over 128 M-tiles), so the per-tile staging in
// the GEMM becomes pure contiguous f16 block copies.

__global__ void __launch_bounds__(256)
wtrans_kernel(const float* __restrict__ W, _Float16* __restrict__ WT,
              int K, int N) {
  __shared__ float tile[32][33];
  const int kb = blockIdx.y * 32, nb = blockIdx.x * 32;
  const int tx = threadIdx.x & 31, ty = threadIdx.x >> 5;
#pragma unroll
  for (int i = ty; i < 32; i += 8)
    tile[i][tx] = W[(size_t)(kb + i) * N + nb + tx];
  __syncthreads();
#pragma unroll
  for (int i = ty; i < 32; i += 8)
    WT[(size_t)(nb + i) * K + kb + tx] = (_Float16)tile[tx][i];
}

// ----------------------------------------------------------- patch embed ----
// One block per token (b*64+s); only the 64 kept conv outputs are computed.
// Emits f16 directly (A-operand of the fuse GEMM).

__global__ void __launch_bounds__(256)
patch_kernel(const float* __restrict__ x,
             const float* __restrict__ wS, const float* __restrict__ bS,
             const float* __restrict__ wM, const float* __restrict__ bM,
             const float* __restrict__ wL, const float* __restrict__ bL,
             _Float16* __restrict__ tok) {
  __shared__ float pS[48];    // [ci][4][4]
  __shared__ float pM[192];   // [ci][8][8]
  __shared__ float pL[768];   // [ci][16][16]
  const int bs = blockIdx.x;
  const int b = bs >> 6, s = bs & 63;
  const int t = threadIdx.x;
  const int oLr = (s >> 3) * 16, oLc = (s & 7) * 16;   // stride-16, p=s
  const int oMr = (s >> 2) * 8,  oMc = (s & 3) * 32;   // stride-8,  p=4s
  const int oSr = (s >> 1) * 4,  oSc = (s & 1) * 64;   // stride-4,  p=16s
  const float* xb = x + (size_t)b * 3 * 128 * 128;
  for (int i = t; i < 768; i += 256) {
    int ci = i >> 8, rr = (i >> 4) & 15, cc = i & 15;
    pL[i] = xb[(size_t)(ci * 128 + oLr + rr) * 128 + oLc + cc];
  }
  if (t < 192) {
    int ci = t >> 6, rr = (t >> 3) & 7, cc = t & 7;
    pM[t] = xb[(size_t)(ci * 128 + oMr + rr) * 128 + oMc + cc];
  }
  if (t < 48) {
    int ci = t >> 4, rr = (t >> 2) & 3, cc = t & 3;
    pS[t] = xb[(size_t)(ci * 128 + oSr + rr) * 128 + oSc + cc];
  }
  __syncthreads();
  for (int c = t; c < 1344; c += 256) {
    float acc;
    if (c < 192) {
      acc = bS[c];
      const float* w = wS + (size_t)c * 48;
#pragma unroll 8
      for (int j = 0; j < 48; j++) acc += pS[j] * w[j];
    } else if (c < 576) {
      int co = c - 192;
      acc = bM[co];
      const float* w = wM + (size_t)co * 192;
#pragma unroll 8
      for (int j = 0; j < 192; j++) acc += pM[j] * w[j];
    } else {
      int co = c - 576;
      acc = bL[co];
      const float* w = wL + (size_t)co * 768;
#pragma unroll 8
      for (int j = 0; j < 768; j++) acc += pL[j] * w[j];
    }
    tok[(size_t)bs * 1344 + c] = (_Float16)acc;
  }
}

// ------------------------------------------------------------- WMMA GEMM ----
// C[M,N] = epilogue(A_f16[M,K] @ WT_f16[N,K]^T + bias).
// 128x128 C-tile, 256 threads (8 waves x 16 rows x 128 cols). K in 32-wide
// chunks; tiles are contiguous f16 so staging is pure 16B copies — async
// global->LDS with ping-pong double buffering when available.

template <class TO, bool GELU_ACT, bool ADD_RES, bool ADD_POS>
__global__ void __launch_bounds__(256)
gemm_f16(const _Float16* __restrict__ A, const _Float16* __restrict__ WT,
         const float* __restrict__ bias, const float* __restrict__ res,
         const float* __restrict__ pos, TO* __restrict__ C, int N, int K) {
#if GEMM_ASYNC
  __shared__ __align__(16) _Float16 As[2][128][40];  // +8 halfs pad
  __shared__ __align__(16) _Float16 Bs[2][128][40];
#else
  __shared__ __align__(16) _Float16 As[1][128][40];
  __shared__ __align__(16) _Float16 Bs[1][128][40];
#endif
  const int t = threadIdx.x;
  const int wv = t >> 5, lane = t & 31, lh = lane >> 4, lrow = lane & 15;
  const int mTile = blockIdx.y * 128, nTile = blockIdx.x * 128;
  const int sr = t >> 1, sh = (t & 1) * 16;  // staging: row, 16-half segment
  const _Float16* gA = A + (size_t)(mTile + sr) * K + sh;
  const _Float16* gB = WT + (size_t)(nTile + sr) * K + sh;

  v8f acc[8];
#pragma unroll
  for (int c = 0; c < 8; c++) acc[c] = zero_v8f();

#if GEMM_ASYNC
  auto issue = [&](int buf, int kc) {
    async16(gA + kc,     &As[buf][sr][sh]);
    async16(gA + kc + 8, &As[buf][sr][sh + 8]);
    async16(gB + kc,     &Bs[buf][sr][sh]);
    async16(gB + kc + 8, &Bs[buf][sr][sh + 8]);
  };
  issue(0, 0);
  int p = 0;
  for (int kc = 0; kc < K; kc += 32) {
    __builtin_amdgcn_s_wait_asynccnt(0);  // my copies into buf p landed
    __syncthreads();                      // everyone's did
    if (kc + 32 < K) issue(p ^ 1, kc + 32);
    v16h a = load_frag_a(&As[p][wv * 16 + lrow][0], lh);
#pragma unroll
    for (int c = 0; c < 8; c++) {
      v16h b = load_frag_b(&Bs[p][c * 16 + lrow][0], lh);
      acc[c] = __builtin_amdgcn_wmma_f32_16x16x32_f16(
          false, a, false, b, (short)0, acc[c], false, false);
    }
    p ^= 1;
  }
#else
  for (int kc = 0; kc < K; kc += 32) {
    v8h a0 = *(const v8h*)(gA + kc);
    v8h a1 = *(const v8h*)(gA + kc + 8);
    v8h b0 = *(const v8h*)(gB + kc);
    v8h b1 = *(const v8h*)(gB + kc + 8);
    *(v8h*)(&As[0][sr][sh]) = a0;
    *(v8h*)(&As[0][sr][sh + 8]) = a1;
    *(v8h*)(&Bs[0][sr][sh]) = b0;
    *(v8h*)(&Bs[0][sr][sh + 8]) = b1;
    __syncthreads();
    v16h a = load_frag_a(&As[0][wv * 16 + lrow][0], lh);
#pragma unroll
    for (int c = 0; c < 8; c++) {
      v16h b = load_frag_b(&Bs[0][c * 16 + lrow][0], lh);
      acc[c] = __builtin_amdgcn_wmma_f32_16x16x32_f16(
          false, a, false, b, (short)0, acc[c], false, false);
    }
    __syncthreads();
  }
#endif
  // epilogue: bias (+gelu) (+pos) (+residual), store fp32 or f16
#pragma unroll
  for (int c = 0; c < 8; c++) {
    const int nCol = nTile + c * 16 + lrow;
    const float bv = bias[nCol];
#pragma unroll
    for (int i = 0; i < 8; i++) {
      const int mRow = mTile + wv * 16 + lh * 8 + i;
      float v = acc[c][i] + bv;
      if (GELU_ACT) v = gelu_f(v);
      if (ADD_POS)  v += pos[(mRow & 63) * HID + nCol];
      if (ADD_RES)  v += res[(size_t)mRow * N + nCol];
      C[(size_t)mRow * N + nCol] = (TO)v;
    }
  }
}

// ---------------------------------------------------------------- LayerNorm -
// One wave per token row (768 = 24 * 32 lanes), eps = 1e-12. Output type is
// templated: f16 when feeding a GEMM A-operand, f32 for the final output.

template <class TO>
__global__ void __launch_bounds__(256)
ln_kernel(const float* __restrict__ x, const float* __restrict__ g,
          const float* __restrict__ bta, TO* __restrict__ o) {
  const int t = threadIdx.x, wv = t >> 5, ln = t & 31;
  const int row = blockIdx.x * 8 + wv;
  const float* xr = x + (size_t)row * HID;
  float v[24];
  float s = 0.f;
#pragma unroll
  for (int j = 0; j < 24; j++) { v[j] = xr[ln + 32 * j]; s += v[j]; }
#pragma unroll
  for (int m = 16; m > 0; m >>= 1) s += __shfl_xor(s, m, 32);
  const float mean = s * (1.f / 768.f);
  float s2 = 0.f;
#pragma unroll
  for (int j = 0; j < 24; j++) { float d = v[j] - mean; s2 += d * d; }
#pragma unroll
  for (int m = 16; m > 0; m >>= 1) s2 += __shfl_xor(s2, m, 32);
  const float inv = rsqrtf(s2 * (1.f / 768.f) + 1e-12f);
  TO* orow = o + (size_t)row * HID;
#pragma unroll
  for (int j = 0; j < 24; j++) {
    int cidx = ln + 32 * j;
    orow[cidx] = (TO)((v[j] - mean) * inv * g[cidx] + bta[cidx]);
  }
}

// ------------------------------------------------------- linear attention ---
// One 128-thread block (4 waves) per (batch, head). S = D = 64.
// qk = q@k, num = qk@v (contracting q's feature dim with k/v's position dim,
// faithful to the reference's S==D math). WMMA f16, f32 accumulate.
// Output written as f16 (A-operand of the proj GEMM).

__global__ void __launch_bounds__(128)
attn_kernel(const float* __restrict__ qkv, _Float16* __restrict__ attout,
            float* __restrict__ qkcol) {
  __shared__ __align__(16) _Float16 qs[64][72];   // q   [s][d]
  __shared__ __align__(16) _Float16 kts[64][72];  // k^T [d][s]
  __shared__ __align__(16) _Float16 vts[64][72];  // v^T [e][s]
  __shared__ __align__(16) _Float16 qks[64][72];  // qk  [s][e]
  __shared__ float ksum[64];
  __shared__ float den[64];
  __shared__ float colsum[64];

  const int b = blockIdx.x / NH, h = blockIdx.x % NH;
  const int t = threadIdx.x;
  const int wv = t >> 5, lane = t & 31, lh = lane >> 4, lrow = lane & 15;

  if (t < 64) colsum[t] = 0.f;

  {  // stage q (elu+1), k (elu+1, transposed), v (transposed) as f16
    const int s = t >> 1, d0 = (t & 1) * 32;
    const float* base = qkv + (size_t)(b * SEQ + s) * (3 * HID) + h * HD;
#pragma unroll 8
    for (int j = 0; j < 32; j++) {
      const int d = d0 + j;
      float qv = base[d];
      float kv = base[HID + d];
      float vv = base[2 * HID + d];
      qs[s][d]  = (_Float16)(qv > 0.f ? qv + 1.f : expf(qv));  // elu(x)+1
      kts[d][s] = (_Float16)(kv > 0.f ? kv + 1.f : expf(kv));
      vts[d][s] = (_Float16)vv;
    }
  }
  __syncthreads();
  if (t < 64) {  // ksum[d] = sum_s k[s,d]
    float s = 0.f;
#pragma unroll 8
    for (int j = 0; j < 64; j++) s += (float)kts[t][j];
    ksum[t] = s;
  }
  // ---- qk = q @ k ----
  v8f acc[4];
#pragma unroll
  for (int c = 0; c < 4; c++) acc[c] = zero_v8f();
#pragma unroll
  for (int kc = 0; kc < 64; kc += 32) {
    v16h a = load_frag_a(&qs[wv * 16 + lrow][kc], lh);
#pragma unroll
    for (int c = 0; c < 4; c++) {
      v16h bf = load_frag_b(&kts[c * 16 + lrow][kc], lh);
      acc[c] = __builtin_amdgcn_wmma_f32_16x16x32_f16(
          false, a, false, bf, (short)0, acc[c], false, false);
    }
  }
#pragma unroll
  for (int c = 0; c < 4; c++) {
    float ps = 0.f;
#pragma unroll
    for (int i = 0; i < 8; i++) {
      float v = acc[c][i];
      qks[wv * 16 + lh * 8 + i][c * 16 + lrow] = (_Float16)v;
      ps += v;
    }
    ps += __shfl_xor(ps, 16, 32);
    if (lh == 0) atomicAdd(&colsum[c * 16 + lrow], ps);
  }
  __syncthreads();
  if (t < 64) {  // den[s] = sum_d q[s,d]*ksum[d]
    float s = 0.f;
#pragma unroll 8
    for (int j = 0; j < 64; j++) s += (float)qs[t][j] * ksum[j];
    den[t] = s;
  } else {
    int e = t - 64;  // export per-(b,h) qk column sums for imp
    qkcol[(size_t)(b * NH + h) * 64 + e] = colsum[e];
  }
  __syncthreads();
  // ---- num = qk @ v ; attn = num / (den + 1e-6) ----
  v8f acc2[4];
#pragma unroll
  for (int c = 0; c < 4; c++) acc2[c] = zero_v8f();
#pragma unroll
  for (int kc = 0; kc < 64; kc += 32) {
    v16h a = load_frag_a(&qks[wv * 16 + lrow][kc], lh);
#pragma unroll
    for (int c = 0; c < 4; c++) {
      v16h bf = load_frag_b(&vts[c * 16 + lrow][kc], lh);
      acc2[c] = __builtin_amdgcn_wmma_f32_16x16x32_f16(
          false, a, false, bf, (short)0, acc2[c], false, false);
    }
  }
#pragma unroll
  for (int c = 0; c < 4; c++) {
    const int e = c * 16 + lrow;
#pragma unroll
    for (int i = 0; i < 8; i++) {
      const int m = wv * 16 + lh * 8 + i;
      attout[(size_t)(b * SEQ + m) * HID + h * HD + e] =
          (_Float16)(acc2[c][i] / (den[m] + 1e-6f));
    }
  }
}

// -------------------------------------------------------------- imp head ----
// imp = softmax_e( (1/S) * sum_h qkcol[b,h,e] ); accumulate mean over layers.

__global__ void __launch_bounds__(64)
imp_kernel(const float* __restrict__ qkcol, float* __restrict__ impout) {
  __shared__ float sh[64];
  const int b = blockIdx.x, e = threadIdx.x;
  float s = 0.f;
#pragma unroll
  for (int h = 0; h < NH; h++) s += qkcol[(size_t)(b * NH + h) * 64 + e];
  s *= (1.f / 64.f);
  sh[e] = s;
  __syncthreads();
  for (int off = 32; off > 0; off >>= 1) {
    if (e < off) sh[e] = fmaxf(sh[e], sh[e + off]);
    __syncthreads();
  }
  const float mx = sh[0];
  __syncthreads();
  const float ex = expf(s - mx);
  sh[e] = ex;
  __syncthreads();
  for (int off = 32; off > 0; off >>= 1) {
    if (e < off) sh[e] += sh[e + off];
    __syncthreads();
  }
  impout[(size_t)b * 64 + e] += ex / sh[0] * (1.f / 6.f);
}

// ------------------------------------------------------------------- host ---

extern "C" void kernel_launch(void* const* d_in, const int* in_sizes, int n_in,
                              void* d_out, int out_size, void* d_ws,
                              size_t ws_size, hipStream_t stream) {
  (void)in_sizes; (void)n_in; (void)out_size; (void)ws_size;
  const float* px     = (const float*)d_in[0];
  const float* wS     = (const float*)d_in[1];
  const float* bS     = (const float*)d_in[2];
  const float* wM     = (const float*)d_in[3];
  const float* bM     = (const float*)d_in[4];
  const float* wL     = (const float*)d_in[5];
  const float* bL     = (const float*)d_in[6];
  const float* fuse_w = (const float*)d_in[7];
  const float* fuse_b = (const float*)d_in[8];
  const float* pos    = (const float*)d_in[9];
  const float* norm_g = (const float*)d_in[10 + 6 * 12];
  const float* norm_b = (const float*)d_in[11 + 6 * 12];
  auto LP = [&](int l, int j) { return (const float*)d_in[10 + l * 12 + j]; };

  // workspace carve (all sub-buffers 16B aligned by construction)
  char* w = (char*)d_ws;
  float*    bx     = (float*)w;    w += (size_t)MTOT * HID * 4;       // x (residual stream, fp32)
  _Float16* bh16   = (_Float16*)w; w += (size_t)MTOT * HID * 2;       // LN output (f16 A-operand)
  _Float16* batt16 = (_Float16*)w; w += (size_t)MTOT * HID * 2;       // attention output (f16)
  float*    qkcol  = (float*)w;    w += (size_t)BATCH * NH * 64 * 4;  // per-(b,h) qk col sums
  _Float16* wt16   = (_Float16*)w; w += (size_t)3072 * 768 * 2;       // transposed f16 weights
  char*     bigc   = w;  // shared scratch: tok f16 / qkv f32 / mlp-hidden f16
  _Float16* tok16  = (_Float16*)bigc;
  float*    qkvf   = (float*)bigc;
  _Float16* mlph16 = (_Float16*)bigc;

  float* out_f   = (float*)d_out;                 // [256,64,768]
  float* out_imp = out_f + (size_t)MTOT * HID;    // [256,64]

  zero_kernel<<<MTOT / 256, 256, 0, stream>>>(out_imp, MTOT);

  // patch embed -> tok f16 [16384,1344]
  patch_kernel<<<MTOT, 256, 0, stream>>>(px, wS, bS, wM, bM, wL, bL, tok16);

  // x = gelu(tok @ fuse_w + fuse_b) + pos
  wtrans_kernel<<<dim3(HID / 32, 1344 / 32), 256, 0, stream>>>(fuse_w, wt16, 1344, HID);
  gemm_f16<float, true, false, true>
      <<<dim3(HID / 128, MTOT / 128), 256, 0, stream>>>(
          tok16, wt16, fuse_b, nullptr, pos, bx, HID, 1344);

  for (int l = 0; l < 6; l++) {
    ln_kernel<_Float16><<<MTOT / 8, 256, 0, stream>>>(bx, LP(l, 0), LP(l, 1), bh16);

    wtrans_kernel<<<dim3(3 * HID / 32, HID / 32), 256, 0, stream>>>(LP(l, 2), wt16, HID, 3 * HID);
    gemm_f16<float, false, false, false>
        <<<dim3(3 * HID / 128, MTOT / 128), 256, 0, stream>>>(
            bh16, wt16, LP(l, 3), nullptr, nullptr, qkvf, 3 * HID, HID);

    attn_kernel<<<BATCH * NH, 128, 0, stream>>>(qkvf, batt16, qkcol);
    imp_kernel<<<BATCH, 64, 0, stream>>>(qkcol, out_imp);

    // x = x + attn @ proj_w + proj_b  (in place: each elem read-then-write)
    wtrans_kernel<<<dim3(HID / 32, HID / 32), 256, 0, stream>>>(LP(l, 4), wt16, HID, HID);
    gemm_f16<float, false, true, false>
        <<<dim3(HID / 128, MTOT / 128), 256, 0, stream>>>(
            batt16, wt16, LP(l, 5), bx, nullptr, bx, HID, HID);

    ln_kernel<_Float16><<<MTOT / 8, 256, 0, stream>>>(bx, LP(l, 6), LP(l, 7), bh16);

    wtrans_kernel<<<dim3(4 * HID / 32, HID / 32), 256, 0, stream>>>(LP(l, 8), wt16, HID, 4 * HID);
    gemm_f16<_Float16, true, false, false>
        <<<dim3(4 * HID / 128, MTOT / 128), 256, 0, stream>>>(
            bh16, wt16, LP(l, 9), nullptr, nullptr, mlph16, 4 * HID, HID);

    wtrans_kernel<<<dim3(HID / 32, 4 * HID / 32), 256, 0, stream>>>(LP(l, 10), wt16, 4 * HID, HID);
    gemm_f16<float, false, true, false>
        <<<dim3(HID / 128, MTOT / 128), 256, 0, stream>>>(
            mlph16, wt16, LP(l, 11), bx, nullptr, bx, HID, 4 * HID);
  }

  ln_kernel<float><<<MTOT / 8, 256, 0, stream>>>(bx, norm_g, norm_b, out_f);
}
